// CasualAttention_738734375173
// MI455X (gfx1250) — compile-verified, matrix operand order checked by reading, not run on hardware
//
#include <hip/hip_runtime.h>
#include <hip/hip_bf16.h>
#include <math.h>

// ---------------- constants ----------------
#define BATCH 4
#define SEQ   2048
#define DIM   1024
#define MROWS (BATCH * SEQ)   // 8192

typedef _Float16 half_t;
typedef __attribute__((ext_vector_type(8)))  _Float16 v8h;
typedef __attribute__((ext_vector_type(16))) _Float16 v16h;
typedef __attribute__((ext_vector_type(8)))  float    v8f;

// ---------------- WMMA helpers ----------------
// Load a 16x32 f16 fragment from a row-major matrix (row stride ld, in elems).
// Per ISA 16-bit A/B layout: lane l (m = l&15, hi = l>>4) holds row m,
// K = {hi*8 .. hi*8+7} and {16+hi*8 .. 16+hi*8+7}  -> two b128 loads.
__device__ __forceinline__ v16h load_frag16x32(const half_t* __restrict__ base, int ld) {
    const int lane = threadIdx.x & 31;
    const int m  = lane & 15;
    const int hi = lane >> 4;
    const half_t* p = base + (size_t)m * ld + hi * 8;
    v8h lo = *(const v8h*)(p);
    v8h hv = *(const v8h*)(p + 16);
    return __builtin_shufflevector(lo, hv, 0,1,2,3,4,5,6,7,8,9,10,11,12,13,14,15);
}

__device__ __forceinline__ v8f wmma_f16(v16h a, v16h b, v8f c) {
    return __builtin_amdgcn_wmma_f32_16x16x32_f16(false, a, false, b, (short)0, c, false, false);
}

// ---------------- 64x64 wave microkernel (4x4 sub-tiles of 16x16) ----------------
// A rows at Abase + (16*i)*lda, B(t) rows at Bbase + (16*j)*ldb, k in [0, kmax) step 32.
// ~33 FLOP/B from L2; 16 f32 accumulators (128 VGPRs) + 8 operand frags (64 VGPRs).
// Kernels using this must be __launch_bounds__(32, 1) so the backend allows ~256 VGPRs.
struct Acc64 { v8f a[4][4]; };

__device__ __forceinline__ void mm64x64(const half_t* __restrict__ Abase, int lda,
                                        const half_t* __restrict__ Bbase, int ldb,
                                        int kmax, Acc64& acc) {
#pragma unroll
    for (int i = 0; i < 4; ++i)
#pragma unroll
        for (int j = 0; j < 4; ++j) acc.a[i][j] = (v8f){0,0,0,0,0,0,0,0};

    const int lane = threadIdx.x & 31;
    const int m  = lane & 15;
    const int hi = lane >> 4;

    for (int kk = 0; kk < kmax; kk += 32) {
        // prefetch next k-step (global_prefetch_b8); harmless at the tail
        if (kk + 32 < kmax) {
            __builtin_prefetch(Abase + (size_t)m * lda + kk + 32 + hi * 8, 0, 1);
            __builtin_prefetch(Bbase + (size_t)m * ldb + kk + 32 + hi * 8, 0, 1);
        }
        v16h a[4], b[4];
#pragma unroll
        for (int i = 0; i < 4; ++i)
            a[i] = load_frag16x32(Abase + (size_t)(16 * i) * lda + kk, lda);
#pragma unroll
        for (int j = 0; j < 4; ++j)
            b[j] = load_frag16x32(Bbase + (size_t)(16 * j) * ldb + kk, ldb);
#pragma unroll
        for (int i = 0; i < 4; ++i)
#pragma unroll
            for (int j = 0; j < 4; ++j)
                acc.a[i][j] = wmma_f16(a[i], b[j], acc.a[i][j]);
    }
}

// ---------------- conversion kernels ----------------
__global__ void cvt_x_f16(const float* __restrict__ x, half_t* __restrict__ xb, int n) {
    int i = blockIdx.x * blockDim.x + threadIdx.x;
    if (i < n) xb[i] = (half_t)x[i];
}

// Wt[e*DIM + d] = (half) W[d*DIM + e]   (transpose D_in x D_out -> D_out x D_in)
__global__ void cvt_w_t_f16(const float* __restrict__ W, half_t* __restrict__ Wt) {
    int i = blockIdx.x * blockDim.x + threadIdx.x;   // i = e*DIM + d
    if (i >= DIM * DIM) return;
    int e = i >> 10;
    int d = i & (DIM - 1);
    Wt[i] = (half_t)W[(size_t)d * DIM + e];
}

// ---------------- projection GEMM: Out = A(MxK) * Bt(NxK)^T ----------------
// wave tile 64x64. mode 0: Out half row-major [M, DIM]
// mode 1: Out half transposed per-batch: Out[b*DIM*SEQ + n*SEQ + s], s = row % SEQ
__global__ __launch_bounds__(32, 1)
void gemm_qkv(const half_t* __restrict__ A, const half_t* __restrict__ Bt,
              half_t* __restrict__ Out, int K, int mode) {
    const int m0 = blockIdx.y * 64;
    const int n0 = blockIdx.x * 64;

    Acc64 acc;
    mm64x64(A + (size_t)m0 * K, K, Bt + (size_t)n0 * K, K, K, acc);

    const int lane = threadIdx.x & 31;
    const int col_l = lane & 15;
    const int hi = lane >> 4;
#pragma unroll
    for (int i = 0; i < 4; ++i)
#pragma unroll
        for (int j = 0; j < 4; ++j)
#pragma unroll
            for (int r = 0; r < 8; ++r) {
                int row = m0 + i * 16 + hi * 8 + r;
                int col = n0 + j * 16 + col_l;
                half_t v = (half_t)acc.a[i][j][r];
                if (mode == 0) {
                    Out[(size_t)row * DIM + col] = v;
                } else {
                    int b_ = row >> 11;            // row / SEQ
                    int s  = row & (SEQ - 1);
                    Out[(size_t)b_ * DIM * SEQ + (size_t)col * SEQ + s] = v;
                }
            }
}

// ---------------- scores: attn[b][q][k] = Q[b,q,:] . K[b,k,:]  (raw, f16) ----------------
// grid: x = k-tile (SEQ/64), y = q-tile (SEQ/64), z = batch. Skip tiles above diagonal.
__global__ __launch_bounds__(32, 1)
void scores_kernel(const half_t* __restrict__ Qb, const half_t* __restrict__ Kb,
                   half_t* __restrict__ attn) {
    const int b  = blockIdx.z;
    const int q0 = blockIdx.y * 64;
    const int k0 = blockIdx.x * 64;
    if (k0 > q0 + 63) return;   // uniform per-workgroup: EXEC stays full

    Acc64 acc;
    mm64x64(Qb + (size_t)(b * SEQ + q0) * DIM, DIM,
            Kb + (size_t)(b * SEQ + k0) * DIM, DIM, DIM, acc);

    const int lane = threadIdx.x & 31;
    const int col_l = lane & 15;
    const int hi = lane >> 4;
    half_t* rowbase = attn + (size_t)b * SEQ * SEQ;
#pragma unroll
    for (int i = 0; i < 4; ++i)
#pragma unroll
        for (int j = 0; j < 4; ++j)
#pragma unroll
            for (int r = 0; r < 8; ++r) {
                int q = q0 + i * 16 + hi * 8 + r;
                int k = k0 + j * 16 + col_l;
                rowbase[(size_t)q * SEQ + k] = (half_t)acc.a[i][j][r];
            }
}

// ---------------- softmax (in place, causal): row (b,q), read k<=q, write weights ----------------
__global__ void softmax_kernel(half_t* __restrict__ attn) {
    const int row = blockIdx.x;           // row = b*SEQ + q
    const int q = row & (SEQ - 1);
    half_t* rp = attn + (size_t)row * SEQ;
    const float scale = 0.03125f;         // 1/sqrt(1024)
    const int n = q + 1;

    __shared__ float red[256];

    float mx = -3.402823466e38f;
    for (int k = threadIdx.x; k < n; k += blockDim.x)
        mx = fmaxf(mx, (float)rp[k] * scale);
    red[threadIdx.x] = mx;
    __syncthreads();
    for (int s = 128; s > 0; s >>= 1) {
        if ((int)threadIdx.x < s) red[threadIdx.x] = fmaxf(red[threadIdx.x], red[threadIdx.x + s]);
        __syncthreads();
    }
    mx = red[0];
    __syncthreads();

    float sum = 0.0f;
    for (int k = threadIdx.x; k < n; k += blockDim.x)
        sum += __expf((float)rp[k] * scale - mx);
    red[threadIdx.x] = sum;
    __syncthreads();
    for (int s = 128; s > 0; s >>= 1) {
        if ((int)threadIdx.x < s) red[threadIdx.x] += red[threadIdx.x + s];
        __syncthreads();
    }
    const float inv = 1.0f / red[0];

    for (int k = threadIdx.x; k < n; k += blockDim.x) {
        float w = __expf((float)rp[k] * scale - mx) * inv;
        rp[k] = (half_t)w;
    }
    for (int k = n + threadIdx.x; k < SEQ; k += blockDim.x)
        rp[k] = (half_t)0.0f;
}

// ---------------- output GEMM: out[b,q,e] = sum_k W[b,q,k] * Vt[b,e,k] ----------------
// grid: x = e-tile (DIM/64), y = q-tile (SEQ/64), z = batch. k loop capped at q0+64 (causal).
__global__ __launch_bounds__(32, 1)
void out_gemm(const half_t* __restrict__ attn, const half_t* __restrict__ Vt,
              float* __restrict__ out) {
    const int b  = blockIdx.z;
    const int q0 = blockIdx.y * 64;
    const int e0 = blockIdx.x * 64;
    const int kmax = q0 + 64;             // weights are zero beyond q

    Acc64 acc;
    mm64x64(attn + (size_t)(b * SEQ + q0) * SEQ, SEQ,
            Vt + (size_t)b * DIM * SEQ + (size_t)e0 * SEQ, SEQ, kmax, acc);

    const int lane = threadIdx.x & 31;
    const int col_l = lane & 15;
    const int hi = lane >> 4;
#pragma unroll
    for (int i = 0; i < 4; ++i)
#pragma unroll
        for (int j = 0; j < 4; ++j)
#pragma unroll
            for (int r = 0; r < 8; ++r) {
                int q = q0 + i * 16 + hi * 8 + r;
                int e = e0 + j * 16 + col_l;
                out[((size_t)(b * SEQ + q)) * DIM + e] = acc.a[i][j][r];
            }
}

// ---------------- launch ----------------
extern "C" void kernel_launch(void* const* d_in, const int* in_sizes, int n_in,
                              void* d_out, int out_size, void* d_ws, size_t ws_size,
                              hipStream_t stream) {
    const float* x  = (const float*)d_in[0];
    const float* Wq = (const float*)d_in[1];
    const float* Wk = (const float*)d_in[2];
    const float* Wv = (const float*)d_in[3];
    float* out = (float*)d_out;

    char* ws = (char*)d_ws;
    size_t off = 0;
    half_t* xb  = (half_t*)(ws + off); off += (size_t)MROWS * DIM * 2;       // 16 MB
    half_t* wqt = (half_t*)(ws + off); off += (size_t)DIM * DIM * 2;         //  2 MB
    half_t* wkt = (half_t*)(ws + off); off += (size_t)DIM * DIM * 2;         //  2 MB
    half_t* wvt = (half_t*)(ws + off); off += (size_t)DIM * DIM * 2;         //  2 MB
    half_t* qb  = (half_t*)(ws + off); off += (size_t)MROWS * DIM * 2;       // 16 MB
    half_t* kb  = (half_t*)(ws + off); off += (size_t)MROWS * DIM * 2;       // 16 MB
    half_t* vt  = (half_t*)(ws + off); off += (size_t)MROWS * DIM * 2;       // 16 MB (per-batch transposed)
    half_t* attn= (half_t*)(ws + off); off += (size_t)BATCH * SEQ * SEQ * 2; // 32 MB

    // 1) precision conversion + weight transposes
    {
        int n = MROWS * DIM;
        cvt_x_f16<<<(n + 255) / 256, 256, 0, stream>>>(x, xb, n);
        int nw = DIM * DIM;
        cvt_w_t_f16<<<(nw + 255) / 256, 256, 0, stream>>>(Wq, wqt);
        cvt_w_t_f16<<<(nw + 255) / 256, 256, 0, stream>>>(Wk, wkt);
        cvt_w_t_f16<<<(nw + 255) / 256, 256, 0, stream>>>(Wv, wvt);
    }

    // 2) projections: Q, K row-major; V stored transposed per batch
    {
        dim3 grid(DIM / 64, MROWS / 64);
        gemm_qkv<<<grid, 32, 0, stream>>>(xb, wqt, qb, DIM, 0);
        gemm_qkv<<<grid, 32, 0, stream>>>(xb, wkt, kb, DIM, 0);
        gemm_qkv<<<grid, 32, 0, stream>>>(xb, wvt, vt, DIM, 1);
    }

    // 3) raw scores (lower-triangular tiles only)
    {
        dim3 grid(SEQ / 64, SEQ / 64, BATCH);
        scores_kernel<<<grid, 32, 0, stream>>>(qb, kb, attn);
    }

    // 4) causal softmax in place -> f16 weights
    softmax_kernel<<<BATCH * SEQ, 256, 0, stream>>>(attn);

    // 5) weights @ V
    {
        dim3 grid(DIM / 64, SEQ / 64, BATCH);
        out_gemm<<<grid, 32, 0, stream>>>(attn, vt, out);
    }
}
